// MultiHeadedAttention_59098749993106
// MI455X (gfx1250) — matrix operand order repaired and checked by reading, not compiled, not run on hardware
//
#include <hip/hip_runtime.h>
#include <hip/hip_bf16.h>

// ---------------------------------------------------------------------------
// MI455X (gfx1250) fused multi-head attention.
//   B=4, S=2048, D=1024, H=16, hd=64.
//   d_out = [ out (B,S,D) fp32 | attn (B,H,S,S) fp32 ]
// ---------------------------------------------------------------------------

#define BATCH 4
#define SEQ   2048
#define DMODEL 1024
#define NHEAD 16
#define HDSZ  64

typedef __bf16 v16bf __attribute__((ext_vector_type(16)));
typedef __bf16 v8bf  __attribute__((ext_vector_type(8)));
typedef float  v8f   __attribute__((ext_vector_type(8)));
typedef float  v4f   __attribute__((ext_vector_type(4)));

#if defined(__has_builtin)
#  if __has_builtin(__builtin_amdgcn_global_load_async_to_lds_b128) && \
      __has_builtin(__builtin_amdgcn_s_wait_asynccnt)
#    define HAVE_ASYNC 1
#  endif
#endif
#ifndef HAVE_ASYNC
#  define HAVE_ASYNC 0
#endif

#if HAVE_ASYNC
// Builtin takes int4-vector pointers in explicit address spaces
// (param type per clang: '__attribute__((vector_size(16))) int *').
typedef int int4v __attribute__((vector_size(16)));
typedef __attribute__((address_space(1))) int4v* gptr128;
typedef __attribute__((address_space(3))) int4v* lptr128;
#endif

__device__ __forceinline__ v8f wmma_bf16(v16bf a, v16bf b, v8f c) {
    // D = A(16x32 bf16) * B(32x16 bf16) + C(16x16 f32)
    return __builtin_amdgcn_wmma_f32_16x16x32_bf16(
        false, a, false, b, (short)0, c, false, false);
}

__device__ __forceinline__ v16bf bfcat(v8bf lo, v8bf hi) {
    return __builtin_shufflevector(lo, hi, 0, 1, 2, 3, 4, 5, 6, 7,
                                   8, 9, 10, 11, 12, 13, 14, 15);
}

// A-matrix (16-bit, 16x32) fragment layout (ISA 7.12.2):
//   element e=0..7  -> K = hf*8 + e        (contiguous run)
//   element e=8..15 -> K = 16 + hf*8 + e-8 (contiguous run)
// B-matrix (16-bit, 32x16): K order as A; this code loads A and B both with K = hf*16 + e (contiguous),
//   N = lane&15.   C/D (f32 16x16): M = r + hf*8, N = lane&15.

// ---------------------------------------------------------------------------
// Tiled GEMM with bias:  out[M,N] = X[M,K] @ W[K,N] + bias[N]
// 128 threads (4 waves). Block tile 64x64, K chunk 32.
// INMODE: 0 = X fp32, 1 = X bf16 (async-LDS staged when available).
// OUTMODE: 0 = fp32 row-major, 1 = bf16 row-major,
//          2 = bf16 transposed per batch: out[b][n][s]  (for V)
// ---------------------------------------------------------------------------
template <int INMODE, int OUTMODE>
__global__ __launch_bounds__(128)
void gemm_bias_kernel(const float* __restrict__ Xf,
                      const __bf16* __restrict__ Xb,
                      const float* __restrict__ W,
                      const float* __restrict__ bias,
                      float* __restrict__ outf,
                      __bf16* __restrict__ outb,
                      int M, int K, int N) {
    __shared__ __bf16 Xs[64][40];   // row-major X tile, 80B row stride
    __shared__ __bf16 Wt[64][40];   // TRANSPOSED W tile: Wt[col][kk]

    const int tid  = threadIdx.x;
    const int w    = tid >> 5;
    const int lane = tid & 31;
    const int hf   = lane >> 4;
    const int ln15 = lane & 15;
    const int sr   = tid >> 1;      // staging row / col   (0..63)
    const int sh   = tid & 1;       // staging half (16 elems)

    const int colBase = blockIdx.x * 64;
    const int rowBase = blockIdx.y * 64;

    v8f cc0 = {}, cc1 = {}, cc2 = {}, cc3 = {};

    for (int k0 = 0; k0 < K; k0 += 32) {
        // ---- stage X tile (64 rows x 32 cols), 16 elems per thread ----
        if constexpr (INMODE == 0) {
            const float* gsrc = Xf + (size_t)(rowBase + sr) * K + k0 + sh * 16;
            v8f x0 = *(const v8f*)gsrc;
            v8f x1 = *(const v8f*)(gsrc + 8);
            v8bf y0, y1;
#pragma unroll
            for (int e = 0; e < 8; ++e) { y0[e] = (__bf16)x0[e]; y1[e] = (__bf16)x1[e]; }
            *(v8bf*)&Xs[sr][sh * 16]     = y0;
            *(v8bf*)&Xs[sr][sh * 16 + 8] = y1;
        } else {
            const __bf16* gsrc = Xb + (size_t)(rowBase + sr) * K + k0 + sh * 16;
#if HAVE_ASYNC
            // CDNA5 async LDS DMA (ASYNCcnt-tracked), bypasses VGPRs
            __builtin_amdgcn_global_load_async_to_lds_b128(
                (gptr128)gsrc, (lptr128)&Xs[sr][sh * 16], 0, 0);
            __builtin_amdgcn_global_load_async_to_lds_b128(
                (gptr128)(gsrc + 8), (lptr128)&Xs[sr][sh * 16 + 8], 0, 0);
#else
            *(v8bf*)&Xs[sr][sh * 16]     = *(const v8bf*)gsrc;
            *(v8bf*)&Xs[sr][sh * 16 + 8] = *(const v8bf*)(gsrc + 8);
#endif
        }
        // ---- stage W tile transposed: Wt[col][kk], 16 elems per thread ----
        {
            v8bf w0, w1;
#pragma unroll
            for (int e = 0; e < 8; ++e) {
                w0[e] = (__bf16)W[(size_t)(k0 + sh * 16 + e) * N + colBase + sr];
                w1[e] = (__bf16)W[(size_t)(k0 + sh * 16 + 8 + e) * N + colBase + sr];
            }
            *(v8bf*)&Wt[sr][sh * 16]     = w0;
            *(v8bf*)&Wt[sr][sh * 16 + 8] = w1;
        }
#if HAVE_ASYNC
        if constexpr (INMODE == 1) __builtin_amdgcn_s_wait_asynccnt(0);
#endif
        __syncthreads();

        // ---- A fragment: two contiguous ds_load_b128 runs ----
        const int arow = w * 16 + ln15;
        v16bf a = bfcat(*(const v8bf*)&Xs[arow][hf * 16],
                        *(const v8bf*)&Xs[arow][hf * 16 + 8]);

        // ---- 4 B fragments from transposed LDS (contiguous runs) ----
        v16bf b;
        b = bfcat(*(const v8bf*)&Wt[0 * 16 + ln15][hf * 16],
                  *(const v8bf*)&Wt[0 * 16 + ln15][hf * 16 + 8]);
        cc0 = wmma_bf16(a, b, cc0);
        b = bfcat(*(const v8bf*)&Wt[1 * 16 + ln15][hf * 16],
                  *(const v8bf*)&Wt[1 * 16 + ln15][hf * 16 + 8]);
        cc1 = wmma_bf16(a, b, cc1);
        b = bfcat(*(const v8bf*)&Wt[2 * 16 + ln15][hf * 16],
                  *(const v8bf*)&Wt[2 * 16 + ln15][hf * 16 + 8]);
        cc2 = wmma_bf16(a, b, cc2);
        b = bfcat(*(const v8bf*)&Wt[3 * 16 + ln15][hf * 16],
                  *(const v8bf*)&Wt[3 * 16 + ln15][hf * 16 + 8]);
        cc3 = wmma_bf16(a, b, cc3);

        __syncthreads();
    }

    // ---- epilogue: bias add + store ----
    v8f accs[4] = {cc0, cc1, cc2, cc3};
#pragma unroll
    for (int j = 0; j < 4; ++j) {
#pragma unroll
        for (int r = 0; r < 8; ++r) {
            int grow = rowBase + w * 16 + r + hf * 8;
            int gcol = colBase + j * 16 + ln15;
            float v  = accs[j][r] + bias[gcol];
            if constexpr (OUTMODE == 0) {
                outf[(size_t)grow * N + gcol] = v;
            } else if constexpr (OUTMODE == 1) {
                outb[(size_t)grow * N + gcol] = (__bf16)v;
            } else {
                int b2 = grow >> 11;          // grow / SEQ
                int s2 = grow & (SEQ - 1);    // grow % SEQ
                outb[((size_t)b2 * N + gcol) * SEQ + s2] = (__bf16)v;
            }
        }
    }
}

// ---------------------------------------------------------------------------
// Fused causal attention for one (batch, head, 16-query-row tile).
// 128 threads (4 waves). Dynamic LDS: 16 x SEQ fp32 score rows (128 KB of
// the CDNA5 320KB WGP LDS). attn streamed to HBM exactly once; P@V runs
// from LDS probabilities; V is read pre-transposed (contiguous fragments).
// ---------------------------------------------------------------------------
__global__ __launch_bounds__(128)
void attention_kernel(const __bf16* __restrict__ Qp,
                      const __bf16* __restrict__ Kp,
                      const __bf16* __restrict__ Vt,   // [B][D][S]
                      __bf16* __restrict__ Ctx,
                      float* __restrict__ attn_out) {
    extern __shared__ float sc[];            // [16][SEQ]
    __shared__ float red[16][8];
    __shared__ float rowmax[16];
    __shared__ float rowsum[16];

    const int tid  = threadIdx.x;
    const int w    = tid >> 5;
    const int lane = tid & 31;
    const int hf   = lane >> 4;
    const int ln15 = lane & 15;

    const int qt = blockIdx.x;
    const int h  = blockIdx.y;
    const int bb = blockIdx.z;
    const int q0 = qt * 16;
    const int nkt = qt + 1;                  // causal: key tiles 0..qt
    const int L   = nkt * 16;                // valid key length
    const int L2  = (L + 31) & ~31;          // zero-padded to WMMA K chunk

    // ---------------- Q fragments (contiguous 16B runs) --------------------
    const __bf16* qbase = Qp + ((size_t)(bb * SEQ + q0)) * DMODEL + h * HDSZ
                             + (size_t)ln15 * DMODEL;
    v16bf aq0 = bfcat(*(const v8bf*)(qbase + hf * 16),
                      *(const v8bf*)(qbase + hf * 16 + 8));
    v16bf aq1 = bfcat(*(const v8bf*)(qbase + 32 + hf * 16),
                      *(const v8bf*)(qbase + 32 + hf * 16 + 8));

    // ---------------- scores: S = (Q K^T)/sqrt(hd), causal-masked ----------
    for (int kt = w; kt < nkt; kt += 4) {    // wave-uniform loop
        const __bf16* kbase = Kp + ((size_t)(bb * SEQ + kt * 16)) * DMODEL
                                 + h * HDSZ + (size_t)ln15 * DMODEL;
        v16bf b0 = bfcat(*(const v8bf*)(kbase + hf * 16),
                         *(const v8bf*)(kbase + hf * 16 + 8));
        v16bf b1 = bfcat(*(const v8bf*)(kbase + 32 + hf * 16),
                         *(const v8bf*)(kbase + 32 + hf * 16 + 8));
        v8f c = {};
        c = wmma_bf16(aq0, b0, c);
        c = wmma_bf16(aq1, b1, c);
#pragma unroll
        for (int r = 0; r < 8; ++r) {
            int m  = r + hf * 8;             // local query row 0..15
            int kg = kt * 16 + ln15;         // global key index
            float sv = (kg <= q0 + m) ? c[r] * 0.125f : -1.0e30f;
            sc[m * SEQ + kg] = sv;
        }
    }
    __syncthreads();

    // ---------------- softmax over [0, L) per row --------------------------
    {
        const int m  = tid >> 3;             // 8 threads per row
        const int sl = tid & 7;
        float pm = -1.0e30f;
        for (int k = sl * 4; k < L; k += 32) {
            v4f x = *(const v4f*)&sc[m * SEQ + k];
            pm = fmaxf(pm, fmaxf(fmaxf(x[0], x[1]), fmaxf(x[2], x[3])));
        }
        red[m][sl] = pm;
        __syncthreads();
        if (sl == 0) {
            float rm = red[m][0];
#pragma unroll
            for (int i = 1; i < 8; ++i) rm = fmaxf(rm, red[m][i]);
            rowmax[m] = rm;
        }
        __syncthreads();
        float rm = rowmax[m];
        float ps = 0.0f;
        for (int k = sl; k < L; k += 8) {
            float e = __expf(sc[m * SEQ + k] - rm);
            sc[m * SEQ + k] = e;
            ps += e;
        }
        red[m][sl] = ps;
        __syncthreads();
        if (sl == 0) {
            float rs = red[m][0];
#pragma unroll
            for (int i = 1; i < 8; ++i) rs += red[m][i];
            rowsum[m] = rs;
        }
        __syncthreads();
    }

    // ------- normalize in LDS + stream attn to HBM (float4, once) ----------
    float* attn = attn_out + (((size_t)bb * NHEAD + h) * SEQ + q0) * SEQ;
    for (int m = 0; m < 16; ++m) {
        const float inv = 1.0f / rowsum[m];
        for (int k = tid * 4; k < SEQ; k += 512) {
            v4f p = {0.f, 0.f, 0.f, 0.f};
            if (k < L) {                      // L,k multiples of 16/4: no split
                v4f x = *(const v4f*)&sc[m * SEQ + k];
                p = x * inv;
                *(v4f*)&sc[m * SEQ + k] = p;
            } else if (k < L2) {
                *(v4f*)&sc[m * SEQ + k] = p;  // zero-pad tail for PV chunks
            }
            *(v4f*)&attn[(size_t)m * SEQ + k] = p;
        }
    }
    __syncthreads();

    // ---------------- ctx = P @ V  (LDS probs x transposed V) --------------
    const int n0 = w * 16;                   // each wave: 16 of 64 hd columns
    const __bf16* vtb = Vt + ((size_t)bb * DMODEL + h * HDSZ + n0 + ln15) * SEQ;
    v8f cc = {};
    const int nk2 = (L + 31) >> 5;
    for (int kt2 = 0; kt2 < nk2; ++kt2) {    // wave-uniform, no predication
        const float* prow = &sc[ln15 * SEQ + kt2 * 32];
        v8f flo = *(const v8f*)(prow + hf * 16);
        v8f fhi = *(const v8f*)(prow + hf * 16 + 8);
        v16bf a;
#pragma unroll
        for (int e = 0; e < 8; ++e) { a[e] = (__bf16)flo[e]; a[e + 8] = (__bf16)fhi[e]; }
        const __bf16* vrun = vtb + kt2 * 32 + hf * 16;
        v16bf b = bfcat(*(const v8bf*)vrun, *(const v8bf*)(vrun + 8));
        cc = wmma_bf16(a, b, cc);
    }
#pragma unroll
    for (int r = 0; r < 8; ++r) {
        int q = q0 + r + hf * 8;
        Ctx[((size_t)(bb * SEQ + q)) * DMODEL + h * HDSZ + n0 + ln15] = (__bf16)cc[r];
    }
}

// ---------------------------------------------------------------------------
extern "C" void kernel_launch(void* const* d_in, const int* in_sizes, int n_in,
                              void* d_out, int out_size, void* d_ws, size_t ws_size,
                              hipStream_t stream) {
    (void)in_sizes; (void)n_in; (void)out_size; (void)ws_size;

    const float* queries = (const float*)d_in[0];
    const float* keys    = (const float*)d_in[1];
    const float* values  = (const float*)d_in[2];
    const float* W_Q = (const float*)d_in[3];
    const float* b_Q = (const float*)d_in[4];
    const float* W_K = (const float*)d_in[5];
    const float* b_K = (const float*)d_in[6];
    const float* W_V = (const float*)d_in[7];
    const float* b_V = (const float*)d_in[8];
    const float* W_O = (const float*)d_in[9];
    const float* b_O = (const float*)d_in[10];

    float* outf = (float*)d_out;                               // (B,S,D)
    float* attn = outf + (size_t)BATCH * SEQ * DMODEL;         // (B,H,S,S)

    const size_t NE = (size_t)BATCH * SEQ * DMODEL;            // 8,388,608
    __bf16* Qp  = (__bf16*)d_ws;
    __bf16* Kp  = Qp + NE;
    __bf16* Vt  = Kp + NE;     // transposed per batch: [B][D][S]
    __bf16* Ctx = Vt + NE;                                     // total ~67 MB

    const int M = BATCH * SEQ;        // 8192
    const int K = DMODEL;             // 1024
    const int N = DMODEL;             // 1024
    dim3 ggrid(N / 64, M / 64);       // (16, 128)
    dim3 gblock(128);

    // Q/K projections: fp32 -> bf16 row-major
    gemm_bias_kernel<0, 1><<<ggrid, gblock, 0, stream>>>(
        queries, nullptr, W_Q, b_Q, nullptr, Qp, M, K, N);
    gemm_bias_kernel<0, 1><<<ggrid, gblock, 0, stream>>>(
        keys, nullptr, W_K, b_K, nullptr, Kp, M, K, N);
    // V projection: fp32 -> bf16 transposed [B][D][S]
    gemm_bias_kernel<0, 2><<<ggrid, gblock, 0, stream>>>(
        values, nullptr, W_V, b_V, nullptr, Vt, M, K, N);

    // Fused causal attention (softmax + attn writeout + P@V)
    dim3 agrid(SEQ / 16, NHEAD, BATCH);                        // (128,16,4)
    size_t lds = (size_t)16 * SEQ * sizeof(float);             // 128 KB
    attention_kernel<<<agrid, dim3(128), lds, stream>>>(Qp, Kp, Vt, Ctx, attn);

    // Output projection: bf16 ctx (async-LDS staged) -> fp32 d_out
    gemm_bias_kernel<1, 0><<<ggrid, gblock, 0, stream>>>(
        nullptr, Ctx, W_O, b_O, outf, nullptr, M, K, N);
}
